// Transformer_8151847928255
// MI455X (gfx1250) — matrix operand order repaired and checked by reading, not compiled, not run on hardware
//
#include <hip/hip_runtime.h>
#include <hip/hip_bf16.h>
#include <stdint.h>

#define USE_ASYNC_COPY 1   // gfx1250 GLOBAL_LOAD_ASYNC_TO_LDS path (ASYNCcnt)

typedef __attribute__((ext_vector_type(16))) _Float16 v16h;
typedef __attribute__((ext_vector_type(8)))  _Float16 v8h;
typedef __attribute__((ext_vector_type(8)))  float    v8f;

union V16U { v16h v; v8h h8[2]; };

__device__ __forceinline__ float gelu_exact(float v) {
    return 0.5f * v * (1.0f + erff(v * 0.70710678118654752f));
}

// ---------------------------------------------------------------------------
// fp32 [K][N] -> f16 [N][K] tiled transpose-convert (weights, once per launch)
// ---------------------------------------------------------------------------
__global__ __launch_bounds__(256)
void cvt_transpose_f16(const float* __restrict__ in, _Float16* __restrict__ out,
                       int Kdim, int Ndim) {
    __shared__ float tile[32][33];
    const int tx = threadIdx.x & 31;
    const int ty = threadIdx.x >> 5;          // 0..7
    const int n0 = blockIdx.x * 32;
    const int k0 = blockIdx.y * 32;
#pragma unroll
    for (int j = 0; j < 32; j += 8)
        tile[ty + j][tx] = in[(long long)(k0 + ty + j) * Ndim + n0 + tx];
    __syncthreads();
#pragma unroll
    for (int j = 0; j < 32; j += 8)
        out[(long long)(n0 + ty + j) * Kdim + k0 + tx] = (_Float16)tile[tx][ty + j];
}

// ---------------------------------------------------------------------------
// LayerNorm: one block per row, fp32 input, f16 output
// ---------------------------------------------------------------------------
__global__ __launch_bounds__(256)
void layernorm_f16(const float* __restrict__ x, const float* __restrict__ g,
                   const float* __restrict__ bta, _Float16* __restrict__ y, int Dd) {
    long long row = blockIdx.x;
    const float* xr = x + row * (long long)Dd;
    float s = 0.f, s2 = 0.f;
    for (int i = threadIdx.x; i < Dd; i += 256) { float v = xr[i]; s += v; s2 += v * v; }
    __shared__ float r1[256];
    __shared__ float r2[256];
    r1[threadIdx.x] = s; r2[threadIdx.x] = s2;
    __syncthreads();
    for (int t = 128; t > 0; t >>= 1) {
        if ((int)threadIdx.x < t) { r1[threadIdx.x] += r1[threadIdx.x + t]; r2[threadIdx.x] += r2[threadIdx.x + t]; }
        __syncthreads();
    }
    float mean = r1[0] / (float)Dd;
    float var  = r2[0] / (float)Dd - mean * mean;
    float rstd = rsqrtf(var + 1e-5f);
    _Float16* yr = y + row * (long long)Dd;
    for (int i = threadIdx.x; i < Dd; i += 256)
        yr[i] = (_Float16)((xr[i] - mean) * rstd * g[i] + bta[i]);
}

// ---------------------------------------------------------------------------
// Causal softmax over rows of length T=256 (one block per row), f32 -> f16
// ---------------------------------------------------------------------------
__global__ __launch_bounds__(256)
void softmax_causal(const float* __restrict__ scores, _Float16* __restrict__ att, int T) {
    long long row = blockIdx.x;          // b*H*T + h*T + i
    int i = (int)(row % T);
    int j = threadIdx.x;
    const float* sr = scores + row * (long long)T;
    float v = (j <= i) ? sr[j] : -3.0e38f;
    __shared__ float red[256];
    red[j] = v;
    __syncthreads();
    for (int t = 128; t > 0; t >>= 1) {
        if (j < t) red[j] = fmaxf(red[j], red[j + t]);
        __syncthreads();
    }
    float m = red[0];
    __syncthreads();
    float e = (j <= i) ? __expf(v - m) : 0.f;
    red[j] = e;
    __syncthreads();
    for (int t = 128; t > 0; t >>= 1) {
        if (j < t) red[j] += red[j + t];
        __syncthreads();
    }
    float ssum = red[0];
    att[row * (long long)T + j] = (_Float16)(e / ssum);
}

// ---------------------------------------------------------------------------
// Tiled WMMA GEMM, double-buffered LDS, async global->LDS staging.
//   C = epilogue(scale * A*B^T + bias [+ resid]);  B is ALWAYS [N][K].
//   Block tile 128x128x32, 256 threads = 8 waves (4x2), wave = 2x4 WMMA tiles.
//   EPI: 0 = (bias), 1 = bias+GELU, 2 = bias+residual-add
//   BIASROW: bias indexed by output row (M) instead of column (N)
//   Batch offset for operand X: (z / Hdiv)*sXo + (z % Hdiv)*sXi
// ---------------------------------------------------------------------------
template <int EPI, bool BIASROW, typename OutT>
__global__ __launch_bounds__(256, 1)
void gemm_f16_wmma(const _Float16* __restrict__ A, int lda, long long sAo, long long sAi,
                   const _Float16* __restrict__ Bm, int ldb, long long sBo, long long sBi,
                   OutT* __restrict__ C, int ldc, long long sCo, long long sCi,
                   const float* __restrict__ bias,
                   const float* __restrict__ resid, int ldr, long long sRo, long long sRi,
                   int K, int Hdiv, float scale) {
    const int z  = blockIdx.z;
    const long long zo = z / Hdiv, zi = z % Hdiv;
    A  += zo * sAo + zi * sAi;
    Bm += zo * sBo + zi * sBi;
    C  += zo * sCo + zi * sCi;
    if (EPI == 2) resid += zo * sRo + zi * sRi;

    __shared__ alignas(16) _Float16 As[2][128][40];   // 80-byte rows: b128 aligned
    __shared__ alignas(16) _Float16 Bs[2][128][40];   // stored [N][K]

    const int tid   = threadIdx.x;
    const int lane  = tid & 31;
    const int wave  = tid >> 5;     // 0..7
    const int waveM = wave >> 1;    // 0..3  -> 32 rows each
    const int waveN = wave & 1;     // 0..1  -> 64 cols each
    const int group = lane >> 4;    // 0/1
    const int l16   = lane & 15;

    const int blockM = blockIdx.y * 128;
    const int blockN = blockIdx.x * 128;

    // staging: 2 x 16-byte chunks per thread per operand; byte voffsets < 2^31
    int sr[2], sc[2];
    unsigned vofA[2], vofB[2];
#pragma unroll
    for (int c = 0; c < 2; ++c) {
        int chunk = tid + c * 256;       // 0..511
        sr[c] = chunk >> 2;              // 0..127
        sc[c] = (chunk & 3) * 8;         // 0,8,16,24
        vofA[c] = (unsigned)(((long long)(blockM + sr[c]) * lda + sc[c]) * 2);
        vofB[c] = (unsigned)(((long long)(blockN + sr[c]) * ldb + sc[c]) * 2);
    }
    const unsigned long long abase = (unsigned long long)(uintptr_t)A;
    const unsigned long long bbase = (unsigned long long)(uintptr_t)Bm;

    v8f acc[2][4] = {};

#if USE_ASYNC_COPY
    auto stage = [&](int k0, int buf) {
        const unsigned kb = (unsigned)(k0 * 2);
#pragma unroll
        for (int c = 0; c < 2; ++c) {
            asm volatile("global_load_async_to_lds_b128 %0, %1, %2 offset:0"
                         :
                         : "v"((unsigned)(uintptr_t)&As[buf][sr[c]][sc[c]]),
                           "v"(vofA[c] + kb), "s"(abase)
                         : "memory");
            asm volatile("global_load_async_to_lds_b128 %0, %1, %2 offset:0"
                         :
                         : "v"((unsigned)(uintptr_t)&Bs[buf][sr[c]][sc[c]]),
                           "v"(vofB[c] + kb), "s"(bbase)
                         : "memory");
        }
    };
    auto stage_wait = [&]() { asm volatile("s_wait_asynccnt 0" ::: "memory"); };
#else
    v8h a_reg[2], b_reg[2];
    auto stage = [&](int k0, int buf) {
#pragma unroll
        for (int c = 0; c < 2; ++c) {
            a_reg[c] = *(const v8h*)(A + (long long)(blockM + sr[c]) * lda + k0 + sc[c]);
            b_reg[c] = *(const v8h*)(Bm + (long long)(blockN + sr[c]) * ldb + k0 + sc[c]);
        }
#pragma unroll
        for (int c = 0; c < 2; ++c) {
            *(v8h*)&As[buf][sr[c]][sc[c]] = a_reg[c];
            *(v8h*)&Bs[buf][sr[c]][sc[c]] = b_reg[c];
        }
    };
    auto stage_wait = [&]() {};
#endif

    auto compute = [&](int buf) {
        V16U afrag[2], bfrag[4];
        const int m0 = waveM * 32;
        const int n0 = waveN * 64;
#pragma unroll
        for (int t = 0; t < 2; ++t) {
            int r = m0 + t * 16 + l16;
            afrag[t].h8[0] = *(const v8h*)&As[buf][r][8 * group];       // K = 8g..8g+7
            afrag[t].h8[1] = *(const v8h*)&As[buf][r][16 + 8 * group];  // K = 16+8g..
        }
#pragma unroll
        for (int u = 0; u < 4; ++u) {
            int n = n0 + u * 16 + l16;
            bfrag[u].h8[0] = *(const v8h*)&Bs[buf][n][8 * group];       // K = 8g..8g+7
            bfrag[u].h8[1] = *(const v8h*)&Bs[buf][n][16 + 8 * group];
        }
#pragma unroll
        for (int t = 0; t < 2; ++t)
#pragma unroll
            for (int u = 0; u < 4; ++u)
                acc[t][u] = __builtin_amdgcn_wmma_f32_16x16x32_f16(
                    false, afrag[t].v, false, bfrag[u].v,
                    (short)0, acc[t][u], false, false);
    };

    // ---- software pipeline: async-stage k+32 while computing k ----
    stage(0, 0);
    stage_wait();
    __syncthreads();
    int cur = 0;
    for (int k0 = 32; k0 < K; k0 += 32) {
        stage(k0, cur ^ 1);                          // ASYNCcnt loads in flight
        if (k0 + 32 < K) {                           // hint tile two steps ahead
            __builtin_prefetch(A + (long long)(blockM + sr[0]) * lda + k0 + 32, 0, 1);
            __builtin_prefetch(Bm + (long long)(blockN + sr[0]) * ldb + k0 + 32, 0, 1);
        }
        compute(cur);                                // WMMA from current buffer
        stage_wait();                                // s_wait_asynccnt 0
        __syncthreads();
        cur ^= 1;
    }
    compute(cur);

    // ---- epilogue: acc element r -> M = tile + r + 8*group, N = tile + l16 ----
#pragma unroll
    for (int t = 0; t < 2; ++t) {
#pragma unroll
        for (int u = 0; u < 4; ++u) {
            const int ncol = blockN + waveN * 64 + u * 16 + l16;
            const float bn = (!BIASROW && bias) ? bias[ncol] : 0.f;
#pragma unroll
            for (int r = 0; r < 8; ++r) {
                const int mrow = blockM + waveM * 32 + t * 16 + 8 * group + r;
                float v = acc[t][u][r] * scale + bn;
                if (BIASROW) v += bias[mrow];
                if (EPI == 1) v = gelu_exact(v);
                if (EPI == 2) v += resid[(long long)mrow * ldr + ncol];
                C[(long long)mrow * ldc + ncol] = (OutT)v;
            }
        }
    }
}

// ---------------------------------------------------------------------------
// Host-side orchestration
// ---------------------------------------------------------------------------
extern "C" void kernel_launch(void* const* d_in, const int* in_sizes, int n_in,
                              void* d_out, int out_size, void* d_ws, size_t ws_size,
                              hipStream_t stream) {
    constexpr int Bb = 32, T = 256, D = 2048, H = 8, Dh = 256;
    constexpr int BT = Bb * T;          // 8192
    constexpr int F  = 4 * D;           // 8192
    constexpr long long DD  = (long long)D * D;
    constexpr long long DF  = (long long)D * F;

    const float* x     = (const float*)d_in[0];
    const float* ln1_g = (const float*)d_in[1];
    const float* ln1_b = (const float*)d_in[2];
    const float* wq    = (const float*)d_in[3];
    const float* bq    = (const float*)d_in[4];
    const float* wk    = (const float*)d_in[5];
    const float* bk    = (const float*)d_in[6];
    const float* wv    = (const float*)d_in[7];
    const float* bv    = (const float*)d_in[8];
    const float* wo    = (const float*)d_in[9];
    const float* bo    = (const float*)d_in[10];
    const float* ln2_g = (const float*)d_in[11];
    const float* ln2_b = (const float*)d_in[12];
    const float* w1    = (const float*)d_in[13];
    const float* b1    = (const float*)d_in[14];
    const float* w2    = (const float*)d_in[15];
    const float* b2    = (const float*)d_in[16];
    float* out = (float*)d_out;

    char* base = (char*)d_ws;
    size_t off = 0;
    auto carve = [&](size_t bytes) -> void* {
        void* p = base + off;
        off += (bytes + 255) & ~(size_t)255;
        return p;
    };

    // transposed f16 weights: [out_dim][in_dim]
    _Float16* wqT    = (_Float16*)carve(DD * 2);
    _Float16* wkT    = (_Float16*)carve(DD * 2);
    _Float16* wvT    = (_Float16*)carve(DD * 2);
    _Float16* woT    = (_Float16*)carve(DD * 2);
    _Float16* w1T    = (_Float16*)carve(DF * 2);
    _Float16* w2T    = (_Float16*)carve(DF * 2);
    _Float16* h16    = (_Float16*)carve((long long)BT * D * 2);
    _Float16* q16    = (_Float16*)carve((long long)BT * D * 2);
    _Float16* k16    = (_Float16*)carve((long long)BT * D * 2);
    _Float16* vT16   = (_Float16*)carve((long long)BT * D * 2);   // [D][BT]
    float*    scores = (float*)   carve((long long)Bb * H * T * T * 4);
    _Float16* att16  = (_Float16*)carve((long long)Bb * H * T * T * 2);
    _Float16* ao16   = (_Float16*)carve((long long)BT * D * 2);
    float*    x1     = (float*)   carve((long long)BT * D * 4);
    _Float16* h216   = (_Float16*)carve((long long)BT * D * 2);
    _Float16* ff16   = (_Float16*)carve((long long)BT * F * 2);

    // 1) weights fp32 [K][N] -> f16 [N][K]
    cvt_transpose_f16<<<dim3(D / 32, D / 32), 256, 0, stream>>>(wq, wqT, D, D);
    cvt_transpose_f16<<<dim3(D / 32, D / 32), 256, 0, stream>>>(wk, wkT, D, D);
    cvt_transpose_f16<<<dim3(D / 32, D / 32), 256, 0, stream>>>(wv, wvT, D, D);
    cvt_transpose_f16<<<dim3(D / 32, D / 32), 256, 0, stream>>>(wo, woT, D, D);
    cvt_transpose_f16<<<dim3(F / 32, D / 32), 256, 0, stream>>>(w1, w1T, D, F);
    cvt_transpose_f16<<<dim3(D / 32, F / 32), 256, 0, stream>>>(w2, w2T, F, D);

    // 2) ln1
    layernorm_f16<<<BT, 256, 0, stream>>>(x, ln1_g, ln1_b, h16, D);

    // 3) Q,K projections: [BT,D] x [D,D]^T + bias -> f16 [BT,D]
    dim3 gProj(D / 128, BT / 128, 1);
    gemm_f16_wmma<0, false, _Float16><<<gProj, 256, 0, stream>>>(
        h16, D, 0, 0, wqT, D, 0, 0, q16, D, 0, 0, bq, nullptr, 0, 0, 0, D, 1, 1.0f);
    gemm_f16_wmma<0, false, _Float16><<<gProj, 256, 0, stream>>>(
        h16, D, 0, 0, wkT, D, 0, 0, k16, D, 0, 0, bk, nullptr, 0, 0, 0, D, 1, 1.0f);

    // 3b) V projection TRANSPOSED: V^T[d,bt] = sum_k wvT[d,k]*h16[bt,k]  (+bv by row)
    dim3 gVT(BT / 128, D / 128, 1);
    gemm_f16_wmma<0, true, _Float16><<<gVT, 256, 0, stream>>>(
        wvT, D, 0, 0, h16, D, 0, 0, vT16, BT, 0, 0, bv, nullptr, 0, 0, 0, D, 1, 1.0f);

    // 4) scores = Q * K^T / sqrt(Dh), batched over (b,h)
    dim3 gScore(T / 128, T / 128, Bb * H);
    gemm_f16_wmma<0, false, float><<<gScore, 256, 0, stream>>>(
        q16, D, (long long)T * D, Dh,
        k16, D, (long long)T * D, Dh,
        scores, T, (long long)H * T * T, (long long)T * T,
        nullptr, nullptr, 0, 0, 0, Dh, H, 0.0625f);

    // 5) causal softmax -> f16
    softmax_causal<<<Bb * H * T, 256, 0, stream>>>(scores, att16, T);

    // 6) att * V = att * (V^T)^T : B operand = V^T head slice [Dh][T], ldb = BT
    dim3 gAV(Dh / 128, T / 128, Bb * H);
    gemm_f16_wmma<0, false, _Float16><<<gAV, 256, 0, stream>>>(
        att16, T, (long long)H * T * T, (long long)T * T,
        vT16, BT, (long long)T, (long long)Dh * BT,
        ao16, D, (long long)T * D, Dh,
        nullptr, nullptr, 0, 0, 0, T, H, 1.0f);

    // 7) out projection + bias + residual(x) -> x1 (fp32)
    gemm_f16_wmma<2, false, float><<<gProj, 256, 0, stream>>>(
        ao16, D, 0, 0, woT, D, 0, 0, x1, D, 0, 0, bo, x, D, 0, 0, D, 1, 1.0f);

    // 8) ln2
    layernorm_f16<<<BT, 256, 0, stream>>>(x1, ln2_g, ln2_b, h216, D);

    // 9) MLP up + GELU -> f16
    dim3 gUp(F / 128, BT / 128, 1);
    gemm_f16_wmma<1, false, _Float16><<<gUp, 256, 0, stream>>>(
        h216, D, 0, 0, w1T, D, 0, 0, ff16, F, 0, 0, b1, nullptr, 0, 0, 0, D, 1, 1.0f);

    // 10) MLP down + bias + residual(x1) -> d_out (fp32)
    dim3 gDown(D / 128, BT / 128, 1);
    gemm_f16_wmma<2, false, float><<<gDown, 256, 0, stream>>>(
        ff16, F, 0, 0, w2T, F, 0, 0, out, D, 0, 0, b2, x1, D, 0, 0, F, 1, 1.0f);
}
